// transformer_attention_40621800686089
// MI455X (gfx1250) — matrix operand order repaired and checked by reading, not compiled, not run on hardware
//
#include <hip/hip_runtime.h>

// ---------------------------------------------------------------------------
// CDNA5 (gfx1250) bf16 WMMA attention layer, double-buffered LDS staging.
// ---------------------------------------------------------------------------

typedef __attribute__((ext_vector_type(16))) __bf16 v16bf;
typedef __attribute__((ext_vector_type(8)))  __bf16 v8bf;
typedef __attribute__((ext_vector_type(4)))  __bf16 v4bf;
typedef __attribute__((ext_vector_type(8)))  float  v8f;

#define WMMA_BF16(a, b, c) \
  __builtin_amdgcn_wmma_f32_16x16x32_bf16(false, (a), false, (b), (short)0, (c), false, false)

// A-matrix 16x32 bf16 fragment: lane m = lane&15, koff = (lane>>4)*8.
// Elements e<8 -> K = koff+e ; e>=8 -> K = koff+16+(e-8).
static __device__ __forceinline__ v16bf frag_a16(const __bf16* p) {
  v8bf lo = *(const v8bf*)(p);
  v8bf hi = *(const v8bf*)(p + 16);
  return __builtin_shufflevector(lo, hi, 0,1,2,3,4,5,6,7,8,9,10,11,12,13,14,15);
}
// B-matrix 32x16 bf16 fragment: lane n = lane&15, koff = (lane>>4)*8, same K split as A.
static __device__ __forceinline__ v16bf frag_b16(const __bf16* p) {
  v8bf lo = *(const v8bf*)(p);
  v8bf hi = *(const v8bf*)(p + 16);
  return __builtin_shufflevector(lo, hi, 0,1,2,3,4,5,6,7,8,9,10,11,12,13,14,15);
}

// ---------------------------------------------------------------------------
// fp32 -> bf16 conversion (float4 -> 4x bf16)
// ---------------------------------------------------------------------------
__global__ __launch_bounds__(256) void cvt_bf16_kernel(const float* __restrict__ in,
                                                       __bf16* __restrict__ out, int n) {
  int i = blockIdx.x * blockDim.x + threadIdx.x;
  int stride = gridDim.x * blockDim.x;
  const int n4 = n >> 2;
  for (; i < n4; i += stride) {
    float4 x = ((const float4*)in)[i];
    v4bf o;
    o[0] = (__bf16)x.x; o[1] = (__bf16)x.y; o[2] = (__bf16)x.z; o[3] = (__bf16)x.w;
    ((v4bf*)out)[i] = o;
  }
}

// ---------------------------------------------------------------------------
// GEMM: Y[r,n] = sum_e X[r,e] * W[n,e]   (torch Linear: x @ W.T)
// 128x128 tile per 256-thread (8-wave) block; K-step 32; double-buffered LDS.
// mode 0: store bf16
// mode 1: xpos rotary * t * 1/sqrt(128), store bf16   (Q)
// mode 2: xpos rotary * 1/t, store bf16               (K)
// mode 3: store fp32                                  (output projection)
// mode 4: store bf16 transposed to [b,h][d][s]        (V for attention)
// ---------------------------------------------------------------------------
__global__ __launch_bounds__(256)
void gemm128_kernel(const __bf16* __restrict__ X, const __bf16* __restrict__ W,
                    __bf16* __restrict__ Obf, float* __restrict__ Of,
                    int E, int N, int S, int mode) {
  __shared__ alignas(16) __bf16 As[2][128][40];  // 80B rows: 16B aligned, 20-bank stride
  __shared__ alignas(16) __bf16 Bs[2][128][40];

  const int tid  = threadIdx.x;
  const int w    = tid >> 5;
  const int lane = tid & 31;
  const int ln   = lane & 15;
  const int hb   = lane >> 4;
  const int mbase = blockIdx.y * 128;
  const int nbase = blockIdx.x * 128;

  // staging map: thread copies rows rA and rA+64, 16B chunk cA (of 4 per row)
  const int rA = tid >> 2;
  const int cA = (tid & 3) * 8;
  const __bf16* gA0 = X + (size_t)(mbase + rA) * E + cA;
  const __bf16* gA1 = gA0 + (size_t)64 * E;
  const __bf16* gB0 = W + (size_t)(nbase + rA) * E + cA;
  const __bf16* gB1 = gB0 + (size_t)64 * E;

  uint4 ra0 = *(const uint4*)gA0;
  uint4 ra1 = *(const uint4*)gA1;
  uint4 rb0 = *(const uint4*)gB0;
  uint4 rb1 = *(const uint4*)gB1;
  *(uint4*)&As[0][rA][cA]      = ra0;
  *(uint4*)&As[0][rA + 64][cA] = ra1;
  *(uint4*)&Bs[0][rA][cA]      = rb0;
  *(uint4*)&Bs[0][rA + 64][cA] = rb1;

  v8f acc[8] = {};
  const int nk = E >> 5;

  for (int k = 0; k < nk; ++k) {
    const int cur = k & 1;
    __syncthreads();
    const bool more = (k + 1) < nk;
    if (more) {                           // issue next tile's global loads early
      const int off = (k + 1) << 5;
      ra0 = *(const uint4*)(gA0 + off);
      ra1 = *(const uint4*)(gA1 + off);
      rb0 = *(const uint4*)(gB0 + off);
      rb1 = *(const uint4*)(gB1 + off);
    }
    v16bf a = frag_a16(&As[cur][w * 16 + ln][hb * 8]);
#pragma unroll
    for (int f = 0; f < 8; ++f) {
      v16bf b = frag_b16(&Bs[cur][f * 16 + ln][hb * 8]);
      acc[f] = WMMA_BF16(a, b, acc[f]);
    }
    if (more) {                           // land next tile after compute
      const int nxt = cur ^ 1;
      *(uint4*)&As[nxt][rA][cA]      = ra0;
      *(uint4*)&As[nxt][rA + 64][cA] = ra1;
      *(uint4*)&Bs[nxt][rA][cA]      = rb0;
      *(uint4*)&Bs[nxt][rA + 64][cA] = rb1;
    }
  }

  const float QSCALE = 0.08838834764831845f;  // 1/sqrt(128)
#pragma unroll
  for (int f = 0; f < 8; ++f) {
    const int col = nbase + f * 16 + ln;
#pragma unroll
    for (int r = 0; r < 8; ++r) {
      const int row = mbase + w * 16 + r + hb * 8;   // flat [B*S) row
      float v = acc[f][r];
      if (mode == 1 || mode == 2) {
        // xpos rotary; pair element lives in adjacent lane (n^1)
        float other = __shfl_xor(v, 1, 32);
        int d = col & 127;
        float rot = (d & 1) ? other : -other;        // rotate_every_two
        float drange = 2.0f * (float)((d >> 1) + 1);
        float theta = __expf(drange * (1.0f / 128.0f) * (-9.210340371976184f)); // 1e-4^(dr/128)
        float zeta  = (drange * (1.0f / 64.0f) + 51.2f) * (1.0f / 52.2f);
        float seq   = ((float)(row & (S - 1)) - 0.5f * (float)S) * (1.0f / 512.0f);
        float sn, c;
        __sincosf(seq * theta, &sn, &c);
        float tfac = __expf(((mode == 1) ? seq : -seq) * __logf(zeta)); // zeta^(+/-seq)
        float res = (v * c + rot * sn) * tfac;
        if (mode == 1) res *= QSCALE;
        Obf[(size_t)row * N + col] = (__bf16)res;
      } else if (mode == 0) {
        Obf[(size_t)row * N + col] = (__bf16)v;
      } else if (mode == 4) {
        // V transposed: [(b*16+h)*128 + d][s], contiguous along s per lane
        int bb = row / S, s = row - bb * S;
        int hh = col >> 7, d = col & 127;
        Obf[((size_t)((bb * 16 + hh) * 128 + d)) * S + s] = (__bf16)v;
      } else {
        Of[(size_t)row * N + col] = v;
      }
    }
  }
}

// ---------------------------------------------------------------------------
// Causal flash attention, bf16 WMMA, fp32 online softmax, double-buffered KV.
// grid: (S/128, B*H). 8 waves; wave w owns 16 q-rows. KV blocks of 64 keys.
// V is pre-transposed: Vg[(bh*128 + d)*S + s].
// ---------------------------------------------------------------------------
__global__ __launch_bounds__(256)
void flash_attn_kernel(const __bf16* __restrict__ Q, const __bf16* __restrict__ K,
                       const __bf16* __restrict__ Vg, __bf16* __restrict__ O, int S) {
  __shared__ alignas(16) __bf16 Ks[2][64][136];   // [key][d]  272B rows
  __shared__ alignas(16) __bf16 Vt[2][128][72];   // [d][key]  144B rows
  __shared__ alignas(16) __bf16 Pl[8][16][72];    // per-wave P re-layout patch

  const int tid  = threadIdx.x;
  const int w    = tid >> 5;
  const int lane = tid & 31;
  const int ln   = lane & 15;
  const int hb   = lane >> 4;
  const int b    = blockIdx.y >> 4;
  const int h    = blockIdx.y & 15;
  const int qs0  = blockIdx.x * 128 + w * 16;

  const size_t kvbase = (size_t)(b * S) * 2048 + (size_t)h * 128;           // K rows
  const size_t vbase  = (size_t)blockIdx.y * 128 * (size_t)S;               // Vg rows

  // Q fragments (d in 4 chunks of 32); lane row = qs0+ln
  const size_t rowQ = (size_t)(b * S + qs0 + ln) * 2048 + (size_t)h * 128;
  v16bf aq[4];
#pragma unroll
  for (int c = 0; c < 4; ++c) aq[c] = frag_a16(Q + rowQ + c * 32 + hb * 8);

  v8f accO[8] = {};
  float mrow[8], lrow[8];
#pragma unroll
  for (int r = 0; r < 8; ++r) { mrow[r] = -3.0e38f; lrow[r] = 0.0f; }

  const int nblocks = (blockIdx.x + 1) * 2;

  // staging map: K -> key kKey+16j, 16B chunk kGrp; V -> row vD+32j, chunk vC
  const int kKey = tid >> 4, kGrp = (tid & 15) * 8;
  const int vD   = tid >> 3, vC   = (tid & 7) * 8;
  uint4 rk[4], rv[4];

  auto loadKV = [&](int kb) {
#pragma unroll
    for (int j = 0; j < 4; ++j) {
      rk[j] = *(const uint4*)(K  + kvbase + (size_t)(kb * 64 + kKey + j * 16) * 2048 + kGrp);
      rv[j] = *(const uint4*)(Vg + vbase  + (size_t)(vD + j * 32) * S + kb * 64 + vC);
    }
  };
  auto storeKV = [&](int buf) {
#pragma unroll
    for (int j = 0; j < 4; ++j) {
      *(uint4*)&Ks[buf][kKey + j * 16][kGrp] = rk[j];
      *(uint4*)&Vt[buf][vD + j * 32][vC]     = rv[j];
    }
  };

  loadKV(0);
  storeKV(0);

  for (int kb = 0; kb < nblocks; ++kb) {
    const int cur = kb & 1;
    __syncthreads();
    const bool more = (kb + 1) < nblocks;
    if (more) loadKV(kb + 1);

    // scores: 4 key sub-tiles x 4 d-chunks = 16 WMMAs
    v8f sc[4];
#pragma unroll
    for (int t = 0; t < 4; ++t) {
      v8f s = {};
#pragma unroll
      for (int c = 0; c < 4; ++c) {
        v16bf bk = frag_b16(&Ks[cur][t * 16 + ln][c * 32 + hb * 8]);
        s = WMMA_BF16(aq[c], bk, s);
      }
      sc[t] = s;
    }

    // online softmax (C-layout row r: m = r + 8*hb)
#pragma unroll
    for (int r = 0; r < 8; ++r) {
      const int qrow = qs0 + r + hb * 8;
      float mx = -3.0e38f;
#pragma unroll
      for (int t = 0; t < 4; ++t) {
        int keyg = kb * 64 + t * 16 + ln;
        float sv = (keyg <= qrow) ? sc[t][r] : -3.0e38f;
        sc[t][r] = sv;
        mx = fmaxf(mx, sv);
      }
      mx = fmaxf(mx, __shfl_xor(mx, 1, 32));
      mx = fmaxf(mx, __shfl_xor(mx, 2, 32));
      mx = fmaxf(mx, __shfl_xor(mx, 4, 32));
      mx = fmaxf(mx, __shfl_xor(mx, 8, 32));
      float mnew = fmaxf(mrow[r], mx);
      float corr = __expf(mrow[r] - mnew);
      mrow[r] = mnew;
      float rs = 0.0f;
#pragma unroll
      for (int t = 0; t < 4; ++t) {
        int keyg = kb * 64 + t * 16 + ln;
        float p = (keyg <= qrow) ? __expf(sc[t][r] - mnew) : 0.0f;
        rs += p;
        Pl[w][r + hb * 8][t * 16 + ln] = (__bf16)p;
      }
      rs += __shfl_xor(rs, 1, 32);
      rs += __shfl_xor(rs, 2, 32);
      rs += __shfl_xor(rs, 4, 32);
      rs += __shfl_xor(rs, 8, 32);
      lrow[r] = lrow[r] * corr + rs;
#pragma unroll
      for (int f = 0; f < 8; ++f) accO[f][r] *= corr;
    }

    // wave-local LDS RAW fence before re-reading P in A layout
    asm volatile("s_wait_dscnt 0" ::: "memory");

    // O += P(16x64) @ V(64x128)
#pragma unroll
    for (int kk = 0; kk < 2; ++kk) {
      v16bf ap = frag_a16(&Pl[w][ln][kk * 32 + hb * 8]);
#pragma unroll
      for (int f = 0; f < 8; ++f) {
        v16bf bv = frag_b16(&Vt[cur][f * 16 + ln][kk * 32 + hb * 8]);
        accO[f] = WMMA_BF16(ap, bv, accO[f]);
      }
    }

    if (more) storeKV(cur ^ 1);
  }

  // normalize and store bf16 attention output [b,s,h*128+d]
#pragma unroll
  for (int f = 0; f < 8; ++f) {
#pragma unroll
    for (int r = 0; r < 8; ++r) {
      const int srow = qs0 + r + hb * 8;
      float o = accO[f][r] / lrow[r];
      O[(size_t)(b * S + srow) * 2048 + h * 128 + f * 16 + ln] = (__bf16)o;
    }
  }
}

// ---------------------------------------------------------------------------
extern "C" void kernel_launch(void* const* d_in, const int* in_sizes, int n_in,
                              void* d_out, int out_size, void* d_ws, size_t ws_size,
                              hipStream_t stream) {
  const int B = 2, S = 2048, E = 2048;
  const int M = B * S;  // 4096 rows

  const float* X  = (const float*)d_in[0];
  const float* Wq = (const float*)d_in[1];
  const float* Wk = (const float*)d_in[2];
  const float* Wv = (const float*)d_in[3];
  const float* Wo = (const float*)d_in[4];
  float* out = (float*)d_out;

  char* ws = (char*)d_ws;
  size_t off = 0;
  __bf16* Xb = (__bf16*)(ws + off); off += (size_t)M * E * 2;  // reused as attn-out
  __bf16* Wb = (__bf16*)(ws + off); off += (size_t)E * E * 2;  // rotating weight slot
  __bf16* Qb = (__bf16*)(ws + off); off += (size_t)M * E * 2;
  __bf16* Kb = (__bf16*)(ws + off); off += (size_t)M * E * 2;
  __bf16* Vb = (__bf16*)(ws + off); off += (size_t)M * E * 2;  // transposed [bh][d][s]
  __bf16* Ab = Xb;

  const dim3 blk(256);
  const dim3 gcvt(1024);
  const dim3 ggemm(E / 128, M / 128);   // (16, 32)
  const dim3 gflash(S / 128, B * 16);   // (16, 32)

  cvt_bf16_kernel<<<gcvt, blk, 0, stream>>>(X, Xb, M * E);

  cvt_bf16_kernel<<<gcvt, blk, 0, stream>>>(Wq, Wb, E * E);
  gemm128_kernel<<<ggemm, blk, 0, stream>>>(Xb, Wb, Qb, nullptr, E, E, S, 1);

  cvt_bf16_kernel<<<gcvt, blk, 0, stream>>>(Wk, Wb, E * E);
  gemm128_kernel<<<ggemm, blk, 0, stream>>>(Xb, Wb, Kb, nullptr, E, E, S, 2);

  cvt_bf16_kernel<<<gcvt, blk, 0, stream>>>(Wv, Wb, E * E);
  gemm128_kernel<<<ggemm, blk, 0, stream>>>(Xb, Wb, Vb, nullptr, E, E, S, 4);

  flash_attn_kernel<<<gflash, blk, 0, stream>>>(Qb, Kb, Vb, Ab, S);

  cvt_bf16_kernel<<<gcvt, blk, 0, stream>>>(Wo, Wb, E * E);
  gemm128_kernel<<<ggemm, blk, 0, stream>>>(Ab, Wb, nullptr, out, E, E, S, 3);
}
